// SpGAT_75453985456248
// MI455X (gfx1250) — matrix operand order repaired and checked by reading, not compiled, not run on hardware
//
#include <hip/hip_runtime.h>
#include <math.h>

#define NN   10000
#define NE   320000
#define LRS  0.2f

typedef __attribute__((ext_vector_type(16))) __bf16       v16bf;
typedef __attribute__((ext_vector_type(8)))  float        v8f;
typedef __attribute__((ext_vector_type(4)))  unsigned int u32x4;
typedef __attribute__((ext_vector_type(4)))  float        f32x4;

static __device__ __forceinline__ unsigned short f2bf(float f) {
    unsigned int u = __float_as_uint(f);
    unsigned int r = (u + 0x7FFFu + ((u >> 16) & 1u)) >> 16;  // RNE
    return (unsigned short)r;
}

// ---------------- conversion / transpose kernels ----------------

__global__ __launch_bounds__(256) void k_f32bf(const float* __restrict__ s,
                                               unsigned short* __restrict__ d, int n) {
    int i = blockIdx.x * 256 + threadIdx.x;
    if (i < n) d[i] = f2bf(s[i]);
}

// WT[n*fin + k] = bf16(W[k*fout + n])   (pre-transpose weights for contiguous B loads)
__global__ __launch_bounds__(256) void k_wt(const float* __restrict__ W,
                                            unsigned short* __restrict__ WT,
                                            int fin, int fout) {
    int i = blockIdx.x * 256 + threadIdx.x;
    if (i >= fin * fout) return;
    int n = i / fin, k = i % fin;
    WT[i] = f2bf(W[(size_t)k * fout + n]);
}

// ---------------- WMMA GEMM:  C[M x Nc] = A[M x K] * BT[Nc x K]^T ----------------
// One wave per 32x32 output block = 2x2 WMMA tiles; each A/B fragment is reused
// twice, so the K-loop issues 8 global_load_b128 per 4 v_wmma_f32_16x16x32_bf16.
// Operand rows are clamped (EXEC must stay all-ones for WMMA); stores are guarded.
__global__ __launch_bounds__(256) void k_gemm(const unsigned short* __restrict__ A,
                                              const unsigned short* __restrict__ BT,
                                              float* __restrict__ C,
                                              int M, int K, int Nc, int sig) {
    int wave = blockIdx.x * (blockDim.x >> 5) + (threadIdx.x >> 5);
    int lane = threadIdx.x & 31;
    int tilesN = (Nc + 31) >> 5, tilesM = (M + 31) >> 5;
    if (wave >= tilesM * tilesN) return;            // wave-uniform exit
    int tm = (wave / tilesN) << 5, tn = (wave % tilesN) << 5;
    int half = lane >> 4, r = lane & 15;

    int rA0 = tm + r;      if (rA0 > M - 1)  rA0 = M - 1;
    int rA1 = tm + 16 + r; if (rA1 > M - 1)  rA1 = M - 1;
    int rB0 = tn + r;      if (rB0 > Nc - 1) rB0 = Nc - 1;
    int rB1 = tn + 16 + r; if (rB1 > Nc - 1) rB1 = Nc - 1;
    const unsigned short* a0 = A  + (size_t)rA0 * K;
    const unsigned short* a1 = A  + (size_t)rA1 * K;
    const unsigned short* b0 = BT + (size_t)rB0 * K;
    const unsigned short* b1 = BT + (size_t)rB1 * K;

    v8f acc00 = {}, acc01 = {}, acc10 = {}, acc11 = {};
    for (int k0 = 0; k0 < K; k0 += 32) {
        union { v16bf v; u32x4 q[2]; } fa0, fa1, fb0, fb1;
        // A 16x32 layout: lanes<16 hold K=[0..7],[16..23]; lanes>=16 hold K=[8..15],[24..31]
        fa0.q[0] = *(const u32x4*)(a0 + k0 + half * 8);
        fa0.q[1] = *(const u32x4*)(a0 + k0 + 16 + half * 8);
        fa1.q[0] = *(const u32x4*)(a1 + k0 + half * 8);
        fa1.q[1] = *(const u32x4*)(a1 + k0 + 16 + half * 8);
        // B 32x16 layout: lanes<16 hold K=[0..7],[16..23]; lanes>=16 hold K=[8..15],[24..31]
        fb0.q[0] = *(const u32x4*)(b0 + k0 + half * 8);
        fb0.q[1] = *(const u32x4*)(b0 + k0 + 16 + half * 8);
        fb1.q[0] = *(const u32x4*)(b1 + k0 + half * 8);
        fb1.q[1] = *(const u32x4*)(b1 + k0 + 16 + half * 8);
        acc00 = __builtin_amdgcn_wmma_f32_16x16x32_bf16(false, fa0.v, false, fb0.v,
                                                        (short)0, acc00, false, false);
        acc01 = __builtin_amdgcn_wmma_f32_16x16x32_bf16(false, fa0.v, false, fb1.v,
                                                        (short)0, acc01, false, false);
        acc10 = __builtin_amdgcn_wmma_f32_16x16x32_bf16(false, fa1.v, false, fb0.v,
                                                        (short)0, acc10, false, false);
        acc11 = __builtin_amdgcn_wmma_f32_16x16x32_bf16(false, fa1.v, false, fb1.v,
                                                        (short)0, acc11, false, false);
    }

    int col0 = tn + r, col1 = tn + 16 + r;
#pragma unroll
    for (int v = 0; v < 8; ++v) {
        int row0 = tm + v + half * 8;
        int row1 = row0 + 16;
        float v00 = acc00[v], v01 = acc01[v], v10 = acc10[v], v11 = acc11[v];
        if (sig) {
            v00 = 1.0f / (1.0f + __expf(-v00)); v01 = 1.0f / (1.0f + __expf(-v01));
            v10 = 1.0f / (1.0f + __expf(-v10)); v11 = 1.0f / (1.0f + __expf(-v11));
        }
        if (row0 < M) {
            if (col0 < Nc) C[(size_t)row0 * Nc + col0] = v00;
            if (col1 < Nc) C[(size_t)row0 * Nc + col1] = v01;
        }
        if (row1 < M) {
            if (col0 < Nc) C[(size_t)row1 * Nc + col0] = v10;
            if (col1 < Nc) C[(size_t)row1 * Nc + col1] = v11;
        }
    }
}

// ---------------- per-node attention logits: as = h.a[:F], ad = h.a[F:] ----------------
__global__ __launch_bounds__(256) void k_alpha(const float* __restrict__ h,
                                               const float* __restrict__ a,
                                               float* __restrict__ as_, float* __restrict__ ad_,
                                               int n, int F) {
    int wave = blockIdx.x * 8 + (threadIdx.x >> 5);
    int lane = threadIdx.x & 31;
    if (wave >= n) return;
    const float* row = h + (size_t)wave * F;
    float s0 = 0.f, s1 = 0.f;
    for (int f = lane; f < F; f += 32) { float v = row[f]; s0 += v * a[f]; s1 += v * a[F + f]; }
    for (int o = 16; o; o >>= 1) { s0 += __shfl_down(s0, o, 32); s1 += __shfl_down(s1, o, 32); }
    if (lane == 0) { as_[wave] = s0; ad_[wave] = s1; }
}

__global__ __launch_bounds__(256) void k_init(float* __restrict__ m, float* __restrict__ den,
                                              float* __restrict__ agg, int n, int tot) {
    int i = blockIdx.x * 256 + threadIdx.x;
    if (i < tot) agg[i] = 0.f;
    if (i < n) { m[i] = -__builtin_inff(); den[i] = 0.f; }
}

static __device__ __forceinline__ void atomicMaxF(float* a, float v) {
    if (v >= 0.f) atomicMax((int*)a, __float_as_int(v));
    else          atomicMin((unsigned int*)a, __float_as_uint(v));
}

__global__ __launch_bounds__(256) void k_emax(const int* __restrict__ src, const int* __restrict__ dst,
                                              const float* __restrict__ as_, const float* __restrict__ ad_,
                                              float* __restrict__ m, int e) {
    int i = blockIdx.x * 256 + threadIdx.x;
    if (i >= e) return;
    float v = as_[src[i]] + ad_[dst[i]];
    v = v > 0.f ? v : LRS * v;
    atomicMaxF(&m[dst[i]], v);
}

__global__ __launch_bounds__(256) void k_ew(const int* __restrict__ src, const int* __restrict__ dst,
                                            const float* __restrict__ as_, const float* __restrict__ ad_,
                                            const float* __restrict__ m, float* __restrict__ w,
                                            float* __restrict__ den, int e) {
    int i = blockIdx.x * 256 + threadIdx.x;
    if (i >= e) return;
    int d = dst[i];
    float v = as_[src[i]] + ad_[d];
    v = v > 0.f ? v : LRS * v;
    float wv = __expf(v - m[d]);
    w[i] = wv;
    atomicAdd(&den[d], wv);
}

__global__ __launch_bounds__(256) void k_scatter(const int* __restrict__ src, const int* __restrict__ dst,
                                                 const float* __restrict__ w, const float* __restrict__ h,
                                                 float* __restrict__ agg, int e, int Fq) {
    long long i = (long long)blockIdx.x * 256 + threadIdx.x;
    if (i >= (long long)e * Fq) return;
    int ed = (int)(i / Fq), f4 = (int)(i % Fq) * 4;
    int F = Fq * 4;
    float ww = w[ed];
    f32x4 v = *(const f32x4*)(h + (size_t)src[ed] * F + f4);
    float* ar = agg + (size_t)dst[ed] * F + f4;
    atomicAdd(ar + 0, ww * v.x); atomicAdd(ar + 1, ww * v.y);
    atomicAdd(ar + 2, ww * v.z); atomicAdd(ar + 3, ww * v.w);
}

__global__ __launch_bounds__(256) void k_fin(const float* __restrict__ agg, const float* __restrict__ den,
                                             float* __restrict__ out, int tot, int F) {
    int i = blockIdx.x * 256 + threadIdx.x;
    if (i >= tot) return;
    float v = agg[i] / (den[i / F] + 1e-16f);
    out[i] = v > 0.f ? v : __expf(v) - 1.0f;   // ELU
}

// z = normalize((z1+z2)/2), one wave per node (128 features -> 4 per lane)
__global__ __launch_bounds__(256) void k_comb(const float* __restrict__ z1, const float* __restrict__ z2,
                                              float* __restrict__ z, int n) {
    int wave = blockIdx.x * 8 + (threadIdx.x >> 5);
    int lane = threadIdx.x & 31;
    if (wave >= n) return;
    f32x4 a = ((const f32x4*)(z1 + (size_t)wave * 128))[lane];
    f32x4 b = ((const f32x4*)(z2 + (size_t)wave * 128))[lane];
    f32x4 v = (a + b) * 0.5f;
    float ss = v.x * v.x + v.y * v.y + v.z * v.z + v.w * v.w;
    for (int o = 16; o; o >>= 1) ss += __shfl_xor(ss, o, 32);
    float nrm = fmaxf(sqrtf(ss), 1e-12f);
    ((f32x4*)(z + (size_t)wave * 128))[lane] = v * (1.0f / nrm);
}

// ---------------- host driver ----------------

extern "C" void kernel_launch(void* const* d_in, const int* in_sizes, int n_in,
                              void* d_out, int out_size, void* d_ws, size_t ws_size,
                              hipStream_t stream) {
    (void)in_sizes; (void)n_in; (void)out_size; (void)ws_size;
    const float* x  = (const float*)d_in[0];
    const float* Bm = (const float*)d_in[1];
    const int*   ei = (const int*)d_in[2];
    const int* src = ei, * dst = ei + NE;
    const float* W[8]; const float* av[8];
    for (int i = 0; i < 8; ++i) { W[i] = (const float*)d_in[3 + 2 * i]; av[i] = (const float*)d_in[4 + 2 * i]; }
    static const int FI[8] = {512, 256, 256, 256, 128, 256, 128, 256};
    static const int FO[8] = {256, 128, 256, 128, 256, 512, 256, 256};

    // workspace bump allocator
    char* p = (char*)d_ws;
    auto alloc = [&](size_t b) -> void* { void* r = p; p += (b + 255) & ~(size_t)255; return r; };
    unsigned short* WT[8];
    for (int i = 0; i < 8; ++i) WT[i] = (unsigned short*)alloc((size_t)FI[i] * FO[i] * 2);
    unsigned short* abf = (unsigned short*)alloc((size_t)NN * 512 * 2);
    unsigned short* zbf = (unsigned short*)alloc((size_t)NN * 128 * 2);
    float* hscr = (float*)alloc((size_t)NN * 512 * 4);
    float* agg  = (float*)alloc((size_t)NN * 512 * 4);
    float* bufA = (float*)alloc((size_t)NN * 256 * 4);
    float* z1b  = (float*)alloc((size_t)NN * 128 * 4);
    float* z2b  = (float*)alloc((size_t)NN * 128 * 4);
    float* as_  = (float*)alloc((size_t)NN * 4);
    float* ad_  = (float*)alloc((size_t)NN * 4);
    float* m_   = (float*)alloc((size_t)NN * 4);
    float* den_ = (float*)alloc((size_t)NN * 4);
    float* wbuf = (float*)alloc((size_t)NE * 4);

    // output layout
    float* out   = (float*)d_out;
    float* Ap    = out;
    float* zOut  = Ap + (size_t)NN * NN;
    float* xhat  = zOut + (size_t)NN * 128;
    float* Bhat  = xhat + (size_t)NN * 512;
    float* xhat2 = Bhat + (size_t)NN * 256;
    float* Bhat2 = xhat2 + (size_t)NN * 512;

    // convert + transpose weights to bf16
    for (int i = 0; i < 8; ++i) {
        int ne = FI[i] * FO[i];
        k_wt<<<(ne + 255) / 256, 256, 0, stream>>>(W[i], WT[i], FI[i], FO[i]);
    }

    auto gatL = [&](const float* xin, int Fin, int Fout,
                    const unsigned short* WTb, const float* a, float* o) {
        int nA = NN * Fin;
        k_f32bf<<<(nA + 255) / 256, 256, 0, stream>>>(xin, abf, nA);
        int tiles = ((NN + 31) / 32) * ((Fout + 31) / 32);
        k_gemm<<<(tiles + 7) / 8, 256, 0, stream>>>(abf, WTb, hscr, NN, Fin, Fout, 0);
        k_alpha<<<(NN + 7) / 8, 256, 0, stream>>>(hscr, a, as_, ad_, NN, Fout);
        int tot = NN * Fout;
        k_init<<<(tot + 255) / 256, 256, 0, stream>>>(m_, den_, agg, NN, tot);
        k_emax<<<(NE + 255) / 256, 256, 0, stream>>>(src, dst, as_, ad_, m_, NE);
        k_ew<<<(NE + 255) / 256, 256, 0, stream>>>(src, dst, as_, ad_, m_, wbuf, den_, NE);
        long long sc = (long long)NE * (Fout / 4);
        k_scatter<<<(int)((sc + 255) / 256), 256, 0, stream>>>(src, dst, wbuf, hscr, agg, NE, Fout / 4);
        k_fin<<<(tot + 255) / 256, 256, 0, stream>>>(agg, den_, o, tot, Fout);
    };

    // encoder
    gatL(x,    512, 256, WT[0], av[0], bufA);   // h
    gatL(bufA, 256, 128, WT[1], av[1], z1b);    // z1
    gatL(Bm,   256, 256, WT[2], av[2], bufA);   // b (reuses h buffer)
    gatL(bufA, 256, 128, WT[3], av[3], z2b);    // z2
    k_comb<<<(NN + 7) / 8, 256, 0, stream>>>(z1b, z2b, zOut, NN);

    // decoders
    gatL(z1b,  128, 256, WT[4], av[4], bufA);   // t1
    gatL(bufA, 256, 512, WT[5], av[5], xhat);
    gatL(zOut, 128, 256, WT[4], av[4], bufA);   // t2
    gatL(bufA, 256, 512, WT[5], av[5], xhat2);
    gatL(z2b,  128, 256, WT[6], av[6], bufA);   // t3
    gatL(bufA, 256, 256, WT[7], av[7], Bhat);
    gatL(zOut, 128, 256, WT[6], av[6], bufA);   // t4
    gatL(bufA, 256, 256, WT[7], av[7], Bhat2);

    // A_pred = sigmoid(z z^T): both operand fragments are row-major rows of z
    k_f32bf<<<((NN * 128) + 255) / 256, 256, 0, stream>>>(zOut, zbf, NN * 128);
    int tz = ((NN + 31) / 32) * ((NN + 31) / 32);
    k_gemm<<<(tz + 7) / 8, 256, 0, stream>>>(zbf, zbf, Ap, NN, 128, NN, 1);
}
